// TemporalMemory_76287209111794
// MI455X (gfx1250) — compile-verified, matrix operand order checked by reading, not run on hardware
//
#include <hip/hip_runtime.h>

#define C_ 1024
#define K_ 16
#define S_ 32
#define Y_ 128
#define N_ (C_*K_)          // 16384 cells
#define ACT_THR_ 10

typedef int v8i __attribute__((ext_vector_type(8)));

__device__ __forceinline__ int clampN(int i) {
    i = i < 0 ? 0 : i;
    return i > (N_-1) ? (N_-1) : i;
}

// ---------------------------------------------------------------- init / acc
__global__ void tm_init(int* __restrict__ cnt) {
    if (threadIdx.x < 2) cnt[threadIdx.x] = 0;
}

__global__ void tm_acc(const int* __restrict__ cnt, float* __restrict__ out_acc) {
    if (threadIdx.x == 0) {
        int na = cnt[0], np = cnt[1];
        int den = na < 1 ? 1 : na;
        out_acc[0] = (na > 0) ? ((float)np / (float)den) : 1.0f;
    }
}

// ---------------------------------------------------------------- phase 1
// block = one column; 256 threads = 8 waves; wave w owns segments [w*64, w*64+64)
__global__ __launch_bounds__(256)
void tm_phase1(const int*   __restrict__ x,
               const int*   __restrict__ prev,
               const int*   __restrict__ conn,
               const float* __restrict__ vol,
               const float* __restrict__ cons,
               float*       __restrict__ out_na,       // d_out[0..N)
               unsigned char* __restrict__ ws_newact,  // N bytes
               unsigned char* __restrict__ ws_learn,   // N bytes
               unsigned char* __restrict__ ws_bestseg, // N bytes
               int*          __restrict__ ws_cnt)      // 2 ints
{
    __shared__ unsigned char prev_lds[N_];           // 16KB: prev active cells as bytes
    __shared__ int ovl_lds[K_*S_];                   // overlap per segment
    __shared__ int act_lds[K_*S_];                   // active_syn per segment
    __shared__ unsigned char ctile[8][16][Y_];       // per-wave WMMA A staging (connected&presyn)
    __shared__ unsigned char ptile[8][16][Y_];       // per-wave WMMA A staging (presyn)
    __shared__ int cellmax[K_], bestseg_sh[K_], predc_sh[K_];
    __shared__ int s_misc[4];                        // 0:prev_any 1:col_pred 2:burst_k 3:x_active

    const int tid  = threadIdx.x;
    const int col  = blockIdx.x;
    const int lane = tid & 31;
    const int wave = tid >> 5;

    if (tid == 0) s_misc[0] = 0;
    __syncthreads();

    // stage prev_active into LDS bytes; compute prev_any
    unsigned int anyloc = 0;
    #pragma unroll
    for (int it = 0; it < 16; ++it) {
        const int i = it*1024 + tid*4;
        const int4 v = *(const int4*)(prev + i);
        prev_lds[i+0] = (unsigned char)(v.x != 0);
        prev_lds[i+1] = (unsigned char)(v.y != 0);
        prev_lds[i+2] = (unsigned char)(v.z != 0);
        prev_lds[i+3] = (unsigned char)(v.w != 0);
        anyloc |= (unsigned int)(v.x | v.y | v.z | v.w);
    }
    if (anyloc) atomicOr(&s_misc[0], 1);
    __syncthreads();

    v8i bones;
    #pragma unroll
    for (int v = 0; v < 8; ++v) bones[v] = 0x01010101;
    v8i zero;
    #pragma unroll
    for (int v = 0; v < 8; ++v) zero[v] = 0;

    const int row = lane & 15;
    const int hi8 = (lane >> 4) << 3;   // 0 for lanes 0-15, 8 for lanes 16-31

    for (int t = 0; t < 4; ++t) {
        const int segb = wave*64 + t*16;   // 16 consecutive segments -> one WMMA tile
        // ---- gather + stage 0/1 bytes into per-wave tiles (coalesced 16B loads)
        for (int r = 0; r < 16; ++r) {
            const int seg = segb + r;                         // == cell*32 + s
            const long long base = ((long long)col*K_*S_ + (long long)seg) * (long long)Y_;
            const int y0 = lane*4;
            const int4   c4 = *(const int4*)  (conn + base + y0);
            const float4 v4 = *(const float4*)(vol  + base + y0);
            const float4 q4 = *(const float4*)(cons + base + y0);
            unsigned int pw = 0, cw = 0;
#define G1(CX, VX, QX, SH) { int idx = clampN(CX); unsigned int p = prev_lds[idx]; \
        unsigned int cc = p & (unsigned int)((VX > 0.5f) || (QX > 0.5f)); \
        pw |= p << SH; cw |= cc << SH; }
            G1(c4.x, v4.x, q4.x, 0)
            G1(c4.y, v4.y, q4.y, 8)
            G1(c4.z, v4.z, q4.z, 16)
            G1(c4.w, v4.w, q4.w, 24)
#undef G1
            *(unsigned int*)&ctile[wave][r][y0] = cw;
            *(unsigned int*)&ptile[wave][r][y0] = pw;
        }
        // ---- build iu8 A fragments per ISA 16x64 layout; all-ones B => row sums
        v8i aC0, aC1, aP0, aP1;
        #pragma unroll
        for (int v = 0; v < 8; ++v) {
            const int kb = ((v>>1)<<4) + ((v&1)<<2) + hi8;
            aC0[v] = *(const int*)&ctile[wave][row][kb];
            aC1[v] = *(const int*)&ctile[wave][row][kb+64];
            aP0[v] = *(const int*)&ptile[wave][row][kb];
            aP1[v] = *(const int*)&ptile[wave][row][kb+64];
        }
        v8i accC = __builtin_amdgcn_wmma_i32_16x16x64_iu8(false, aC0, false, bones, zero, false, false);
        accC     = __builtin_amdgcn_wmma_i32_16x16x64_iu8(false, aC1, false, bones, accC, false, false);
        v8i accP = __builtin_amdgcn_wmma_i32_16x16x64_iu8(false, aP0, false, bones, zero, false, false);
        accP     = __builtin_amdgcn_wmma_i32_16x16x64_iu8(false, aP1, false, bones, accP, false, false);
        // D col 0: lane 0 holds rows 0-7, lane 16 holds rows 8-15
        if (lane == 0 || lane == 16) {
            const int rb = segb + hi8;
            #pragma unroll
            for (int v = 0; v < 8; ++v) {
                act_lds[rb+v] = accC[v];
                ovl_lds[rb+v] = accP[v];
            }
        }
    }
    __syncthreads();

    // ---- per-cell reduction: best segment (first-max) + predicted flag
    if (tid < K_) {
        const int b = tid * S_;
        int best = 0, bmax = ovl_lds[b];
        int pr = (act_lds[b] >= ACT_THR_);
        for (int s = 1; s < S_; ++s) {
            const int ov = ovl_lds[b+s];
            if (ov > bmax) { bmax = ov; best = s; }
            pr |= (act_lds[b+s] >= ACT_THR_);
        }
        cellmax[tid] = bmax; bestseg_sh[tid] = best; predc_sh[tid] = pr;
    }
    __syncthreads();

    // ---- column reduction: predicted, burst winner (flat argmax order), counters
    if (tid == 0) {
        int cp = 0;
        for (int k = 0; k < K_; ++k) cp |= predc_sh[k];
        int bk = 0, bm = cellmax[0];
        for (int k = 1; k < K_; ++k) if (cellmax[k] > bm) { bm = cellmax[k]; bk = k; }
        const int xa = (x[col] != 0);
        s_misc[1] = cp; s_misc[2] = bk; s_misc[3] = xa;
        if (xa)       atomicAdd(&ws_cnt[0], 1);
        if (xa && cp) atomicAdd(&ws_cnt[1], 1);
    }
    __syncthreads();

    // ---- outputs: new_active, learn flags, best segment per cell
    if (tid < K_) {
        const int pany = s_misc[0], cp = s_misc[1], bk = s_misc[2], xa = s_misc[3];
        const int n = col*K_ + tid;
        const int na = xa && !cp;
        out_na[n]     = na ? 1.0f : 0.0f;
        ws_newact[n]  = (unsigned char)na;
        const int winner = xa && (cp ? predc_sh[tid] : (tid == bk));
        ws_learn[n]   = (unsigned char)(winner && pany);
        ws_bestseg[n] = (unsigned char)bestseg_sh[tid];
    }
}

// ---------------------------------------------------------------- phase 2
__global__ __launch_bounds__(256)
void tm_phase2(const int*   __restrict__ prev,
               const int*   __restrict__ conn,
               const float* __restrict__ vol,
               const float* __restrict__ cons,
               const unsigned char* __restrict__ ws_newact,
               const unsigned char* __restrict__ ws_learn,
               const unsigned char* __restrict__ ws_bestseg,
               float* __restrict__ out_pred)           // d_out[N..2N)
{
    __shared__ unsigned char prev_lds[N_];            // for on-the-fly learned-row recompute
    __shared__ unsigned char na_lds[N_];              // new active cells (gather target)
    __shared__ unsigned char ctile[8][16][Y_];
    __shared__ int predcell[K_];
    __shared__ int ovr_flag[K_], ovr_seg[K_];

    const int tid  = threadIdx.x;
    const int col  = blockIdx.x;
    const int lane = tid & 31;
    const int wave = tid >> 5;

    if (tid < K_) {
        const int n = col*K_ + tid;
        predcell[tid] = 0;
        ovr_flag[tid] = ws_learn[n];
        ovr_seg[tid]  = ws_bestseg[n];
    }
    #pragma unroll
    for (int it = 0; it < 16; ++it) {
        const int i = it*1024 + tid*4;
        const int4 v = *(const int4*)(prev + i);
        prev_lds[i+0] = (unsigned char)(v.x != 0);
        prev_lds[i+1] = (unsigned char)(v.y != 0);
        prev_lds[i+2] = (unsigned char)(v.z != 0);
        prev_lds[i+3] = (unsigned char)(v.w != 0);
    }
    {
        const unsigned int* na32 = (const unsigned int*)ws_newact;
        unsigned int* nal = (unsigned int*)na_lds;
        #pragma unroll
        for (int it = 0; it < 16; ++it) {
            const int j = it*256 + tid;
            nal[j] = na32[j];
        }
    }
    __syncthreads();

    v8i bones;
    #pragma unroll
    for (int v = 0; v < 8; ++v) bones[v] = 0x01010101;
    v8i zero;
    #pragma unroll
    for (int v = 0; v < 8; ++v) zero[v] = 0;

    const int row = lane & 15;
    const int hi8 = (lane >> 4) << 3;

    for (int t = 0; t < 4; ++t) {
        const int segb = wave*64 + t*16;
        for (int r = 0; r < 16; ++r) {
            const int seg  = segb + r;
            const int cell = seg >> 5;
            const int s    = seg & 31;
            const long long base = ((long long)col*K_*S_ + (long long)seg) * (long long)Y_;
            const bool ovr = (ovr_flag[cell] != 0) && (s == ovr_seg[cell]);  // wave-uniform
            const int y0 = lane*4;
            const int4   c4 = *(const int4*)  (conn + base + y0);
            const float4 v4 = *(const float4*)(vol  + base + y0);
            const float4 q4 = *(const float4*)(cons + base + y0);
            unsigned int cw = 0;
#define G2(CX, VX, QX, SH) { int idx = clampN(CX); unsigned int pn = na_lds[idx]; \
        float vv = VX; if (ovr) vv += prev_lds[idx] ? 0.1f : -0.01f; \
        unsigned int cc = pn & (unsigned int)((vv > 0.5f) || (QX > 0.5f)); \
        cw |= cc << SH; }
            G2(c4.x, v4.x, q4.x, 0)
            G2(c4.y, v4.y, q4.y, 8)
            G2(c4.z, v4.z, q4.z, 16)
            G2(c4.w, v4.w, q4.w, 24)
#undef G2
            *(unsigned int*)&ctile[wave][r][y0] = cw;
        }
        v8i a0, a1;
        #pragma unroll
        for (int v = 0; v < 8; ++v) {
            const int kb = ((v>>1)<<4) + ((v&1)<<2) + hi8;
            a0[v] = *(const int*)&ctile[wave][row][kb];
            a1[v] = *(const int*)&ctile[wave][row][kb+64];
        }
        v8i acc = __builtin_amdgcn_wmma_i32_16x16x64_iu8(false, a0, false, bones, zero, false, false);
        acc     = __builtin_amdgcn_wmma_i32_16x16x64_iu8(false, a1, false, bones, acc,  false, false);
        if (lane == 0 || lane == 16) {
            const int rb = segb + hi8;
            #pragma unroll
            for (int v = 0; v < 8; ++v) {
                if (acc[v] >= ACT_THR_) atomicOr(&predcell[(rb+v) >> 5], 1);
            }
        }
    }
    __syncthreads();

    if (tid < K_) {
        out_pred[col*K_ + tid] = predcell[tid] ? 1.0f : 0.0f;
    }
}

// ---------------------------------------------------------------- launcher
extern "C" void kernel_launch(void* const* d_in, const int* in_sizes, int n_in,
                              void* d_out, int out_size, void* d_ws, size_t ws_size,
                              hipStream_t stream) {
    const int*   x    = (const int*)d_in[0];
    const int*   prev = (const int*)d_in[1];
    const int*   conn = (const int*)d_in[2];
    const float* vol  = (const float*)d_in[3];
    const float* cons = (const float*)d_in[4];
    float* out = (float*)d_out;                    // [0..N)=new_active [N..2N)=pred_next [2N]=acc

    unsigned char* ws = (unsigned char*)d_ws;
    int* cnt                  = (int*)ws;          // 2 ints
    unsigned char* ws_newact  = ws + 64;
    unsigned char* ws_learn   = ws + 64 + N_;
    unsigned char* ws_bestseg = ws + 64 + 2*N_;

    tm_init<<<1, 32, 0, stream>>>(cnt);
    tm_phase1<<<dim3(C_), dim3(256), 0, stream>>>(x, prev, conn, vol, cons,
                                                  out, ws_newact, ws_learn, ws_bestseg, cnt);
    tm_acc<<<1, 32, 0, stream>>>(cnt, out + 2*N_);
    tm_phase2<<<dim3(C_), dim3(256), 0, stream>>>(prev, conn, vol, cons,
                                                  ws_newact, ws_learn, ws_bestseg, out + N_);
}
